// ChannelAttention2_35562329211750
// MI455X (gfx1250) — matrix operand order repaired and checked by reading, not compiled, not run on hardware
//
#include <hip/hip_runtime.h>
#include <hip/hip_bf16.h>

typedef __bf16 bf16;
typedef __attribute__((ext_vector_type(16))) __bf16 v16bf;
typedef __attribute__((ext_vector_type(8)))  __bf16 v8bf;
typedef __attribute__((ext_vector_type(4)))  __bf16 v4bf;
typedef __attribute__((ext_vector_type(8)))  float  v8f;
typedef __attribute__((ext_vector_type(8)))  unsigned int v8u;

constexpr int kC    = 192;     // channels
constexpr int kNTok = 64;      // tokens per window
constexpr int kNWin = 4608;    // B * Nw = 8 * 576
constexpr float kScale = 0.20412414523193154f; // (192/8)^-0.5

// ---------------------------------------------------------------------------
// Pre-swizzle weights into WMMA B-fragment layout (bf16), fold K-scale.
//   elem(fi, lane, e):  n = tile_n*16 + (lane&15);  k = kstep*32 + (lane>>4)*16 + e
// w_qkv: 36 n-tiles x 6 k-steps = 216 frags; w_proj: 12 x 6 = 72 frags.
// ---------------------------------------------------------------------------
__global__ void prep_weights(const float* __restrict__ wqkv,
                             const float* __restrict__ wproj,
                             bf16* __restrict__ wqs,
                             bf16* __restrict__ wps) {
  int t    = blockIdx.x * blockDim.x + threadIdx.x;  // 0 .. 147455
  int fi   = t >> 9;
  int r    = t & 511;
  int lane = r >> 4;
  int e    = r & 15;
  int kin  = ((lane >> 4) << 4) + e;
  int ncol = lane & 15;
  if (fi < 216) {
    int nt = fi / 6, ks = fi % 6;
    int n = nt * 16 + ncol;
    int k = ks * 32 + kin;
    float v = wqkv[k * 576 + n];
    if (n >= 192 && n < 384) v *= kScale;          // pre-scale K columns
    wqs[t] = (bf16)v;
  } else {
    int f2 = fi - 216;
    int ct = f2 / 6, ks = f2 % 6;
    int n = ct * 16 + ncol;
    int k = ks * 32 + kin;
    wps[(f2 << 9) + r] = (bf16)wproj[k * kC + n];
  }
}

// A fragment (16x32 bf16, MxK) from row-major storage: two 16B runs per lane.
__device__ __forceinline__ v16bf lds_load_A(const bf16* __restrict__ p0, int ld) {
  const int l = threadIdx.x & 31;
  const bf16* p = p0 + (l & 15) * ld + ((l >> 4) << 4);
  uint4 lo = *(const uint4*)(p);
  uint4 hi = *(const uint4*)(p + 8);
  v8u au = {lo.x, lo.y, lo.z, lo.w, hi.x, hi.y, hi.z, hi.w};
  return __builtin_bit_cast(v16bf, au);
}

// B fragment (32x16 bf16, KxN) from n-major storage: elem(n,k) at p0[n*ld+k].
__device__ __forceinline__ v16bf lds_load_B_nmajor(const bf16* __restrict__ p0, int ld) {
  const int l = threadIdx.x & 31;
  const bf16* p = p0 + (l & 15) * ld + ((l >> 4) << 4);
  return *(const v16bf*)p;
}

// B fragment from pre-swizzled global weights: two b128 loads per lane.
__device__ __forceinline__ v16bf load_wfrag(const bf16* __restrict__ w, int fi) {
  const int l = threadIdx.x & 31;
  return *(const v16bf*)(w + (((size_t)fi << 5) + l) * 16);
}

__device__ __forceinline__ v8bf cvt8(v8f a) {
  v8bf h;
#pragma unroll
  for (int r = 0; r < 8; ++r) h[r] = (bf16)a[r];
  return h;
}

#define WMMA_BF16(a, b, c) \
  __builtin_amdgcn_wmma_f32_16x16x32_bf16(false, (a), false, (b), (short)0, (c), false, false)

// ---------------------------------------------------------------------------
// One workgroup = one 64-token window. 384 threads = 12 waves.
// LDS map (bytes):
//   xt  @ 0      : 64x192 bf16  x tile (token-major); reused as OW in stage 3/4
//   qs  @ 24576  : 64x192 bf16  Q (token-major)
//   kt  @ 49152  : 192x64 bf16  K^T (c-major, pre-scaled)
//   vt  @ 73728  : 192x64 bf16  V^T (d-major)
//   s   @ 98304  : 192x192 f32  S^T (d-major logits / exp values)
//   p   @ 245760 : 192x192 bf16 P (c-major softmax probs)
//   red @ 319488 : 384 f32      softmax reduction scratch
// total = 321024 B  (<= 320 KB WGP LDS)
// ---------------------------------------------------------------------------
__global__ void __launch_bounds__(384)
fused_window_attn(const float* __restrict__ x,
                  const bf16*  __restrict__ wqs,
                  const bf16*  __restrict__ wps,
                  const float* __restrict__ bproj,
                  float*       __restrict__ out) {
  extern __shared__ char smem[];
  bf16*  xt  = (bf16*)(smem);
  bf16*  qs  = (bf16*)(smem + 24576);
  bf16*  kt  = (bf16*)(smem + 49152);
  bf16*  vt  = (bf16*)(smem + 73728);
  float* s   = (float*)(smem + 98304);
  bf16*  p   = (bf16*)(smem + 245760);
  float* red = (float*)(smem + 319488);

  const int tid  = threadIdx.x;
  const int wave = tid >> 5;
  const int lane = tid & 31;
  const int win  = blockIdx.x;
  const int ncol = lane & 15;
  const int rb   = (lane >> 4) << 3;        // D-matrix row base per lane half

  // ---- load x window tile: 64x192 f32 -> bf16, packed 8B stores ----
  {
    const float4* xs = (const float4*)(x + (size_t)win * (kNTok * kC));
#pragma unroll
    for (int j = 0; j < 8; ++j) {
      int idx = tid + j * 384;              // 3072 float4 total
      float4 vv = xs[idx];
      v4bf h = {(bf16)vv.x, (bf16)vv.y, (bf16)vv.z, (bf16)vv.w};
      *(v4bf*)(xt + (idx << 2)) = h;
    }
  }
  __syncthreads();

  // ---- stage 1: QKV = X * Wqkv, two-tile software-pipelined weight stream ----
  {
    const int mt = wave & 3;                // token tile 0..3
    const int g  = wave >> 2;               // 0..2; nt = 3*nn + g (wave-uniform)
    v16bf a[6];
#pragma unroll
    for (int ks = 0; ks < 6; ++ks)
      a[ks] = lds_load_A(xt + (mt * 16) * kC + ks * 32, kC);

    // route D tile to Q / K^T / V^T region; nn is scalar (loop counter) ->
    // scalar branches only.
    auto store_qkv = [&](v8f acc, int nn) {
      const int nt = 3 * nn + g;
      if (nn < 4) {                         // Q, token-major (strided b16)
        bf16* dst = qs + (mt * 16 + rb) * kC + nt * 16 + ncol;
#pragma unroll
        for (int r = 0; r < 8; ++r) dst[r * kC] = (bf16)acc[r];
      } else if (nn < 8) {                  // K^T, c-major, packed 16B
        bf16* dst = kt + ((nt - 12) * 16 + ncol) * kNTok + mt * 16 + rb;
        *(v8bf*)dst = cvt8(acc);
      } else {                              // V^T, d-major, packed 16B
        bf16* dst = vt + ((nt - 24) * 16 + ncol) * kNTok + mt * 16 + rb;
        *(v8bf*)dst = cvt8(acc);
      }
    };

    v16bf b[6], bn[6];
#pragma unroll
    for (int ks = 0; ks < 6; ++ks)
      b[ks] = load_wfrag(wqs, g * 6 + ks);  // tile nn=0

#pragma clang loop unroll(disable)
    for (int t2 = 0; t2 < 6; ++t2) {        // pair of tiles per iteration
      const int nn0 = 2 * t2, nn1 = 2 * t2 + 1;
      // prefetch tile nn1 while computing nn0
#pragma unroll
      for (int ks = 0; ks < 6; ++ks)
        bn[ks] = load_wfrag(wqs, (3 * nn1 + g) * 6 + ks);
      v8f acc0 = {};
#pragma unroll
      for (int ks = 0; ks < 6; ++ks) acc0 = WMMA_BF16(a[ks], b[ks], acc0);
      store_qkv(acc0, nn0);
      // prefetch tile nn0+2 while computing nn1
      if (t2 < 5) {
#pragma unroll
        for (int ks = 0; ks < 6; ++ks)
          b[ks] = load_wfrag(wqs, (3 * (nn0 + 2) + g) * 6 + ks);
      }
      v8f acc1 = {};
#pragma unroll
      for (int ks = 0; ks < 6; ++ks) acc1 = WMMA_BF16(a[ks], bn[ks], acc1);
      store_qkv(acc1, nn1);
    }
  }
  __syncthreads();

  // ---- stage 2: S[c,d] = sum_n K[n,c]*V[n,d]; store S^T (d-major), packed ----
  {
    const int ct = wave;                    // 0..11
    v16bf a0 = lds_load_A(kt + (ct * 16) * kNTok + 0,  kNTok);
    v16bf a1 = lds_load_A(kt + (ct * 16) * kNTok + 32, kNTok);
    for (int dt = 0; dt < 12; ++dt) {
      v8f acc = {};
      acc = WMMA_BF16(a0, lds_load_B_nmajor(vt + (dt * 16) * kNTok + 0,  kNTok), acc);
      acc = WMMA_BF16(a1, lds_load_B_nmajor(vt + (dt * 16) * kNTok + 32, kNTok), acc);
      float* dst = s + (dt * 16 + ncol) * kC + ct * 16 + rb;   // S^T, contiguous in r
      *(v8f*)dst = acc;                     // packed 32B store
    }
  }
  __syncthreads();

  // ---- softmax over d (2 threads per row c); P written c-major bf16 ----
  {
    const int c  = (tid < kC) ? tid : tid - kC;
    const int h  = (tid < kC) ? 0 : 1;
    const int d0 = h * 96;
    float* col = s + c;                     // element d at col[d*kC]
    float mx = -INFINITY;
    for (int d = d0; d < d0 + 96; ++d) mx = fmaxf(mx, col[d * kC]);
    red[tid] = mx;
    __syncthreads();
    const float m2 = fmaxf(red[c], red[c + kC]);
    __syncthreads();                        // all maxes consumed before reuse
    float sum = 0.f;
    for (int d = d0; d < d0 + 96; ++d) {
      float e = __expf(col[d * kC] - m2);
      col[d * kC] = e;                      // own slots only
      sum += e;
    }
    red[tid] = sum;
    __syncthreads();
    const float inv = 1.0f / (red[c] + red[c + kC]);
    bf16* prow = p + c * kC + d0;
    for (int j = 0; j < 96; ++j) prow[j] = (bf16)(col[(d0 + j) * kC] * inv);
  }
  __syncthreads();

  // ---- stage 3: O^T = P * Q^T  ->  OW token-major with packed stores ----
  {
    const int ct = wave;                    // c-tile 0..11
    v16bf ap[6];
#pragma unroll
    for (int ks = 0; ks < 6; ++ks)
      ap[ks] = lds_load_A(p + (ct * 16) * kC + ks * 32, kC);
    for (int tt = 0; tt < 4; ++tt) {        // token tiles
      v8f acc = {};
#pragma unroll
      for (int ks = 0; ks < 6; ++ks)
        acc = WMMA_BF16(ap[ks], lds_load_B_nmajor(qs + (tt * 16) * kC + ks * 32, kC), acc);
      bf16* dst = xt + (tt * 16 + ncol) * kC + ct * 16 + rb;   // OW[token][c]
      *(v8bf*)dst = cvt8(acc);
    }
  }
  __syncthreads();

  // ---- stage 4: Y = OW @ Wproj + b, two-tile pipelined weight stream ----
  {
    const int mt = wave & 3;
    const int g  = wave >> 2;
    v16bf a[6];
#pragma unroll
    for (int ks = 0; ks < 6; ++ks)
      a[ks] = lds_load_A(xt + (mt * 16) * kC + ks * 32, kC);

    auto store_y = [&](v8f acc, int i) {
      const int ct = g + 3 * i;
      const float bias = bproj[ct * 16 + ncol];
      float* dst = out + ((size_t)win * kNTok + mt * 16 + rb) * kC + ct * 16 + ncol;
#pragma unroll
      for (int r = 0; r < 8; ++r) dst[r * kC] = acc[r] + bias;
    };

    v16bf b[6], bn[6];
#pragma unroll
    for (int ks = 0; ks < 6; ++ks)
      b[ks] = load_wfrag(wps, g * 6 + ks);  // tile i=0 (ct=g)

#pragma clang loop unroll(disable)
    for (int t2 = 0; t2 < 2; ++t2) {        // tiles (2*t2, 2*t2+1)
      const int i0 = 2 * t2, i1 = 2 * t2 + 1;
#pragma unroll
      for (int ks = 0; ks < 6; ++ks)
        bn[ks] = load_wfrag(wps, (g + 3 * i1) * 6 + ks);
      v8f acc0 = {};
#pragma unroll
      for (int ks = 0; ks < 6; ++ks) acc0 = WMMA_BF16(a[ks], b[ks], acc0);
      store_y(acc0, i0);
      if (t2 < 1) {
#pragma unroll
        for (int ks = 0; ks < 6; ++ks)
          b[ks] = load_wfrag(wps, (g + 3 * (i0 + 2)) * 6 + ks);
      }
      v8f acc1 = {};
#pragma unroll
      for (int ks = 0; ks < 6; ++ks) acc1 = WMMA_BF16(a[ks], bn[ks], acc1);
      store_y(acc1, i1);
    }
  }
}

extern "C" void kernel_launch(void* const* d_in, const int* in_sizes, int n_in,
                              void* d_out, int out_size, void* d_ws, size_t ws_size,
                              hipStream_t stream) {
  const float* x     = (const float*)d_in[0];
  const float* wqkv  = (const float*)d_in[1];
  const float* wproj = (const float*)d_in[2];
  const float* bproj = (const float*)d_in[3];
  // d_in[4]=H, d_in[5]=W scalars; shapes are compile-time constants here.

  bf16* wqs = (bf16*)d_ws;                  // 216 frags * 512 bf16
  bf16* wps = wqs + 216 * 512;              // 72 frags * 512 bf16 (~295 KB ws)

  prep_weights<<<576, 256, 0, stream>>>(wqkv, wproj, wqs, wps);
  fused_window_attn<<<kNWin, 384, 321024, stream>>>(x, wqs, wps, bproj, (float*)d_out);
}
